// GumbelLatentTyper_19370302505047
// MI455X (gfx1250) — matrix operand order repaired and checked by reading, not compiled, not run on hardware
//
#include <hip/hip_runtime.h>
#include <hip/hip_bf16.h>

// ---------------------------------------------------------------------------
// GumbelLatentTyper eval path for MI455X (gfx1250, wave32, WMMA).
//
// logits = x[32768x1024] @ W^T[1024x512] done as split-3 bf16 WMMA
// (x_hi*W_hi + x_hi*W_lo + x_lo*W_hi), ~fp32-accurate (err ~2^-18 rel),
// using V_WMMA_F32_16X16X32_BF16 (8192 MACs/instr vs 1024 for f32 WMMA).
// fp32->bf16 split of x happens in-register and co-executes with the XDL.
// W is pre-split into bf16 hi/lo arrays in d_ws by a tiny prep kernel.
// ---------------------------------------------------------------------------

#define NUM_VARS 512
#define VAR_DIM  256
#define DIM      1024
#define ROWS     32768            // 8*4096 tokens

static const long OUT_OFF  = 0;
static const long OUT_SZ   = (long)ROWS * VAR_DIM;        //  8388608
static const long SOFT_OFF = OUT_OFF + OUT_SZ;            //  8388608
static const long SOFT_SZ  = (long)ROWS * NUM_VARS;       // 16777216
static const long PERP_OFF = SOFT_OFF + SOFT_SZ;          // 25165824
static const long GUMB_OFF = PERP_OFF + 1;                // 25165825

// d_ws byte layout
static const size_t WS_PART_OFF = 0;                       // 1024*512 f32 = 2 MB
static const size_t WS_COL_OFF  = 0x200000;                // 512 f32
static const size_t WS_WHI_OFF  = 0x210000;                // 512*1024 bf16 = 1 MB
static const size_t WS_WLO_OFF  = 0x310000;                // 512*1024 bf16 = 1 MB

typedef __attribute__((ext_vector_type(8)))  float  v8f;
typedef __attribute__((ext_vector_type(16))) __bf16 v16bf;

// ---------------------------------------------------------------------------
// Kernel 0: split W (f32) into bf16 hi/lo. hi = RNE(f), lo = RNE(f - hi).
// ---------------------------------------------------------------------------
__global__ __launch_bounds__(256) void wsplit_kernel(
    const float* __restrict__ W, __bf16* __restrict__ Whi,
    __bf16* __restrict__ Wlo) {
  const int i = blockIdx.x * 256 + threadIdx.x;
  const float f = W[i];
  const __bf16 h = (__bf16)f;
  Whi[i] = h;
  Wlo[i] = (__bf16)(f - (float)h);
}

// ---------------------------------------------------------------------------
// Kernel 1: logits GEMM, split-3 bf16 WMMA.
// Block = 256 threads = 8 waves. Block b owns M rows [32b, 32b+32).
// Waves 0-3: M-tile 2b,   N cols [128w, 128w+128)  (8 N-tiles each)
// Waves 4-7: M-tile 2b+1, N cols [128(w-4), ...)
//
// A (16x32 bf16): lane<16 (M=lane) holds K k0+0..15 (element s = k0+s) as v16bf
//                 elements 0..15; lane>=16 (M=lane-16) holds +16 (same order as B).
// B (32x16 bf16): lane holds 16 contiguous K at k0 + 16*(lane>=16),
//                 N = lane&15  -> per-lane 32B load from row-major W.
// C/D (16x16 f32): lane<16 -> M=r, lane>=16 -> M=8+r, N = lane&15.
// ---------------------------------------------------------------------------
__global__ __launch_bounds__(256) void gemm_logits_kernel(
    const float* __restrict__ x, const __bf16* __restrict__ Whi,
    const __bf16* __restrict__ Wlo, float* __restrict__ logits) {
  const int tid   = threadIdx.x;
  const int wave  = tid >> 5;
  const int lane  = tid & 31;
  const int half  = lane >> 4;     // 0: low-K half, 1: high-K half
  const int l16   = lane & 15;
  const int mtile = blockIdx.x * 2 + (wave >> 2);
  const int n0    = (wave & 3) * 128;

  const float* xrow = x + (size_t)(mtile * 16 + l16) * DIM;
  // per-lane B base: row n0+l16, K offset half*16
  const size_t wbase = (size_t)(n0 + l16) * DIM + half * 16;
  const __bf16* wh = Whi + wbase;
  const __bf16* wl = Wlo + wbase;

  v8f acc[8];
#pragma unroll
  for (int t = 0; t < 8; ++t) acc[t] = (v8f){};

#pragma unroll 2
  for (int k0 = 0; k0 < DIM; k0 += 32) {
    // ---- load + split A fragment (16 fp32 -> bf16 hi/lo) ----
    const float4* pa0 = (const float4*)(xrow + k0 + half * 16);      // K k0+{0..7}(+16)
    const float4* pa1 = (const float4*)(xrow + k0 + half * 16 + 8);  // K k0+8+{0..7}(+16)
    float af[16];
    {
      float4 c0 = pa0[0], c1 = pa0[1], c2 = pa1[0], c3 = pa1[1];
      af[0] = c0.x;  af[1] = c0.y;  af[2] = c0.z;  af[3] = c0.w;
      af[4] = c1.x;  af[5] = c1.y;  af[6] = c1.z;  af[7] = c1.w;
      af[8] = c2.x;  af[9] = c2.y;  af[10] = c2.z; af[11] = c2.w;
      af[12] = c3.x; af[13] = c3.y; af[14] = c3.z; af[15] = c3.w;
    }
    v16bf a_hi, a_lo;
#pragma unroll
    for (int j = 0; j < 16; ++j) {
      const __bf16 h = (__bf16)af[j];
      a_hi[j] = h;
      a_lo[j] = (__bf16)(af[j] - (float)h);
    }

    // ---- 8 N-tiles: 3 WMMAs each (hi*hi, hi*lo, lo*hi) ----
#pragma unroll
    for (int t = 0; t < 8; ++t) {
      const size_t boff = (size_t)t * 16 * DIM + k0;
      v16bf bh = *(const v16bf*)(wh + boff);
      v16bf bl = *(const v16bf*)(wl + boff);
      v8f a = __builtin_amdgcn_wmma_f32_16x16x32_bf16(
          false, a_lo, false, bh, (short)0, acc[t], false, false);
      a = __builtin_amdgcn_wmma_f32_16x16x32_bf16(
          false, a_hi, false, bl, (short)0, a, false, false);
      acc[t] = __builtin_amdgcn_wmma_f32_16x16x32_bf16(
          false, a_hi, false, bh, (short)0, a, false, false);
    }
  }

  float* lr = logits + (size_t)(mtile * 16 + half * 8) * NUM_VARS + n0 + l16;
#pragma unroll
  for (int t = 0; t < 8; ++t)
#pragma unroll
    for (int r = 0; r < 8; ++r)
      lr[(size_t)r * NUM_VARS + t * 16] = acc[t][r];
}

// ---------------------------------------------------------------------------
// Kernel 2: per-row mask/zero-col0, first-occurrence argmax, stable softmax
// (in place over raw logits in the soft_probs region), one-hot gumbel,
// codebook gather, deterministic per-block avg_probs partial (LDS tree).
// Block = 1024 threads = 32 waves = 32 rows.
// ---------------------------------------------------------------------------
__global__ __launch_bounds__(1024) void row_softmax_kernel(
    const float* __restrict__ mask, const float* __restrict__ codebook,
    float* __restrict__ out, float* __restrict__ soft,
    float* __restrict__ gumbel, float* __restrict__ ws_part) {
  __shared__ float lpart[32 * NUM_VARS];  // 64 KB of the 320 KB WGP LDS

  const int tid  = threadIdx.x;
  const int w    = tid >> 5;
  const int lane = tid & 31;
  const int row  = blockIdx.x * 32 + w;
  const float maskv = mask[row];

  float* srow = soft + (size_t)row * NUM_VARS;

  float v[16];
  {
    const float4* p = (const float4*)(srow + lane * 16);
#pragma unroll
    for (int q = 0; q < 4; ++q) {
      float4 t = p[q];
      v[4 * q + 0] = t.x; v[4 * q + 1] = t.y;
      v[4 * q + 2] = t.z; v[4 * q + 3] = t.w;
    }
  }
  if (lane == 0) v[0] = 0.0f;              // zero_mask on column 0

  float mx = -3.4028234663852886e38f;
  int   idx = 0;
#pragma unroll
  for (int j = 0; j < 16; ++j) {
    v[j] *= maskv;                          // per-token mask
    if (v[j] > mx) { mx = v[j]; idx = lane * 16 + j; }  // first occurrence
  }
#pragma unroll
  for (int d = 16; d > 0; d >>= 1) {        // wave32 butterfly argmax
    float omx  = __shfl_xor(mx, d, 32);
    int   oidx = __shfl_xor(idx, d, 32);
    if (omx > mx || (omx == mx && oidx < idx)) { mx = omx; idx = oidx; }
  }

  float s = 0.0f;
#pragma unroll
  for (int j = 0; j < 16; ++j) { v[j] = __expf(v[j] - mx); s += v[j]; }
#pragma unroll
  for (int d = 16; d > 0; d >>= 1) s += __shfl_xor(s, d, 32);
  const float inv = 1.0f / s;

  // soft_probs (in place) + LDS partial (soft * mask)
  {
    float4* sp = (float4*)(srow + lane * 16);
    float*  lp = lpart + w * NUM_VARS + lane * 16;
#pragma unroll
    for (int q = 0; q < 4; ++q) {
      float4 t;
      t.x = v[4 * q + 0] * inv; t.y = v[4 * q + 1] * inv;
      t.z = v[4 * q + 2] * inv; t.w = v[4 * q + 3] * inv;
      sp[q] = t;
      lp[4 * q + 0] = t.x * maskv; lp[4 * q + 1] = t.y * maskv;
      lp[4 * q + 2] = t.z * maskv; lp[4 * q + 3] = t.w * maskv;
    }
  }

  // gumbel_probs one-hot
  {
    float g[16];
#pragma unroll
    for (int j = 0; j < 16; ++j) g[j] = (lane * 16 + j == idx) ? 1.0f : 0.0f;
    float4* gp = (float4*)(gumbel + (size_t)row * NUM_VARS + lane * 16);
#pragma unroll
    for (int q = 0; q < 4; ++q) {
      float4 t;
      t.x = g[4 * q + 0]; t.y = g[4 * q + 1];
      t.z = g[4 * q + 2]; t.w = g[4 * q + 3];
      gp[q] = t;
    }
  }

  // out = hard_x @ codebook(row0 zeroed) == gather row idx (zero if idx==0)
  {
    float4* op = (float4*)(out + (size_t)row * VAR_DIM + lane * 8);
    if (idx == 0) {
      float4 z = {0.0f, 0.0f, 0.0f, 0.0f};
      op[0] = z; op[1] = z;
    } else {
      const float4* cp = (const float4*)(codebook + (size_t)idx * VAR_DIM + lane * 8);
      op[0] = cp[0]; op[1] = cp[1];
    }
  }

  // deterministic cross-wave tree reduce of the 32 per-row vectors
  __syncthreads();
#pragma unroll
  for (int d2 = 16; d2 > 0; d2 >>= 1) {
    if (w < d2) {
      float*       a = lpart + w * NUM_VARS + lane * 16;
      const float* b = lpart + (w + d2) * NUM_VARS + lane * 16;
#pragma unroll
      for (int j = 0; j < 16; ++j) a[j] += b[j];
    }
    __syncthreads();
  }
  if (tid < NUM_VARS)
    ws_part[(size_t)blockIdx.x * NUM_VARS + tid] = lpart[tid];
}

// ---------------------------------------------------------------------------
// Kernel 3: reduce the 1024 per-block partials per column (fixed-order tree).
// ---------------------------------------------------------------------------
__global__ __launch_bounds__(256) void col_reduce_kernel(
    const float* __restrict__ ws_part, float* __restrict__ ws_col) {
  __shared__ float sh[256];
  const int c = blockIdx.x;
  const int t = threadIdx.x;
  float s = 0.0f;
#pragma unroll
  for (int j = 0; j < 4; ++j)
    s += ws_part[(size_t)(t + j * 256) * NUM_VARS + c];
  sh[t] = s;
  __syncthreads();
  for (int d = 128; d > 0; d >>= 1) {
    if (t < d) sh[t] += sh[t + d];
    __syncthreads();
  }
  if (t == 0) ws_col[c] = sh[0];
}

// ---------------------------------------------------------------------------
// Kernel 4: sum(mask), avg_probs, entropy, perplexity. 1 block x 512 threads.
// ---------------------------------------------------------------------------
__global__ __launch_bounds__(512) void finalize_kernel(
    const float* __restrict__ mask, const float* __restrict__ ws_col,
    float* __restrict__ perp_out) {
  __shared__ float sh[512];
  const int t = threadIdx.x;

  float s = 0.0f;
  for (int i = t; i < ROWS; i += 512) s += mask[i];
  sh[t] = s;
  __syncthreads();
  for (int d = 256; d > 0; d >>= 1) {
    if (t < d) sh[t] += sh[t + d];
    __syncthreads();
  }
  const float total = sh[0];
  __syncthreads();

  const float avg = ws_col[t] / total;
  sh[t] = avg * __logf(avg + 1e-7f);
  __syncthreads();
  for (int d = 256; d > 0; d >>= 1) {
    if (t < d) sh[t] += sh[t + d];
    __syncthreads();
  }
  if (t == 0) *perp_out = __expf(-sh[0]);
}

// ---------------------------------------------------------------------------
extern "C" void kernel_launch(void* const* d_in, const int* in_sizes, int n_in,
                              void* d_out, int out_size, void* d_ws, size_t ws_size,
                              hipStream_t stream) {
  const float* x        = (const float*)d_in[0];
  const float* mask     = (const float*)d_in[1];
  const float* W        = (const float*)d_in[2];
  const float* codebook = (const float*)d_in[3];

  float* outp = (float*)d_out;
  char*  wsb  = (char*)d_ws;

  float* out_gather = outp + OUT_OFF;
  float* soft       = outp + SOFT_OFF;   // also scratch for raw logits
  float* perp       = outp + PERP_OFF;
  float* gumbel     = outp + GUMB_OFF;

  float*  ws_part = (float*)(wsb + WS_PART_OFF);  // 1024*512 f32
  float*  ws_col  = (float*)(wsb + WS_COL_OFF);   // 512 f32
  __bf16* Whi     = (__bf16*)(wsb + WS_WHI_OFF);  // 512*1024 bf16
  __bf16* Wlo     = (__bf16*)(wsb + WS_WLO_OFF);  // 512*1024 bf16

  // 0) split W into bf16 hi/lo
  wsplit_kernel<<<(NUM_VARS * DIM) / 256, 256, 0, stream>>>(W, Whi, Wlo);
  // 1) logits = x @ W^T  (split-3 bf16 WMMA), stored into soft region
  gemm_logits_kernel<<<ROWS / 32, 256, 0, stream>>>(x, Whi, Wlo, soft);
  // 2) per-row softmax/argmax/one-hot/gather + per-block avg partials
  row_softmax_kernel<<<ROWS / 32, 1024, 0, stream>>>(mask, codebook,
                                                     out_gather, soft, gumbel,
                                                     ws_part);
  // 3) column-wise reduce of partials
  col_reduce_kernel<<<NUM_VARS, 256, 0, stream>>>(ws_part, ws_col);
  // 4) perplexity scalar
  finalize_kernel<<<1, 512, 0, stream>>>(mask, ws_col, perp);
}
